// GPTOneLayer_34995393528512
// MI455X (gfx1250) — matrix operand order repaired and checked by reading, not compiled, not run on hardware
//
#include <hip/hip_runtime.h>
#include <cmath>

#define SEQ    2048
#define NVOCAB 50257
#define DMODEL 768
#define NHEADS 8
#define HDIM   96
#define DFF    3072

typedef __attribute__((ext_vector_type(16))) _Float16 v16h;
typedef __attribute__((ext_vector_type(8)))  _Float16 v8h;
typedef __attribute__((ext_vector_type(8)))  float    v8f;

// ---------------------------------------------------------------- elementwise
__global__ void f32_to_f16_kernel(const float* __restrict__ in,
                                  _Float16* __restrict__ out, int n) {
  int i = blockIdx.x * blockDim.x + threadIdx.x;
  if (i < n) out[i] = (_Float16)in[i];
}

// x[s,d] = in_embed[tok[s],d] + pos_embed[s,d]; writes f32 + f16 copies.
__global__ void embed_kernel(const int* __restrict__ tok,
                             const float* __restrict__ emb,
                             const float* __restrict__ pos,
                             float* __restrict__ x32,
                             _Float16* __restrict__ x16) {
  int i = blockIdx.x * blockDim.x + threadIdx.x;  // exact SEQ*DMODEL grid
  int s = i / DMODEL;
  int d = i - s * DMODEL;
  float v = emb[(size_t)tok[s] * DMODEL + d] + pos[(size_t)s * DMODEL + d];
  x32[i] = v;
  x16[i] = (_Float16)v;
}

__device__ __forceinline__ float gelu_exact(float x) {
  return 0.5f * x * (1.0f + erff(x * 0.70710678118654752440f));
}

// ---------------------------------------------------------------- WMMA GEMM
// C[M,N] = act( (A[M,K] * Bt[N,K]^T) * scale + bias )
// One wave computes a 32x64 tile: 2 M-subtiles x 4 N-subtiles,
// 8 x v_wmma_f32_16x16x32_f16 per K-step of 32.
// grid = (ceil(N/64), M/32), block = 32 (wave32).
// Fast path (wave-uniform full tile) has no guards in the inner loop.
template <bool HAS_BIAS, bool DO_GELU, bool STORE32, bool STORE16,
          bool TRANS16, bool CAUSAL_SKIP>
__global__ void __launch_bounds__(32)
wmma_gemm(const _Float16* __restrict__ A, int lda,
          const _Float16* __restrict__ Bt, int ldb,
          const float* __restrict__ bias,
          float* __restrict__ C32, int ldc32,
          _Float16* __restrict__ C16, int ldc16,
          int N, int K, float scale) {
  const int lane = threadIdx.x;
  const int m0 = blockIdx.y * 32;
  const int n0 = blockIdx.x * 64;
  if (CAUSAL_SKIP && n0 > m0 + 31) return;  // tile fully above causal diagonal

  const int r16 = lane & 15;   // row (A) / col (B) within 16
  const int hi  = lane >> 4;   // lane-group select

  // A fragment pointers: lane group 0 holds K {0..15},
  // lane group 1 holds K {16..31} (matches B).
  const _Float16* aPtr[2];
#pragma unroll
  for (int mi = 0; mi < 2; ++mi)
    aPtr[mi] = A + (size_t)(m0 + mi * 16 + r16) * lda + hi * 16;

  // B fragment pointers: lane holds column n, lane group selects contiguous
  // K {0..15} / {16..31} (16 halves = one 32B load).
  const _Float16* bPtr[4];
#pragma unroll
  for (int j = 0; j < 4; ++j) {
    int col = n0 + j * 16 + r16;
    bPtr[j] = Bt + (size_t)(col < N ? col : 0) * ldb + hi * 16;
  }

  v8f acc[2][4] = {};

  const bool fullTile = (n0 + 64 <= N);  // wave-uniform

  if (fullTile) {
    // -------- fast path: no guards, clean load + 8xWMMA inner loop --------
#pragma unroll 2
    for (int k0 = 0; k0 < K; k0 += 32) {
      v16h a[2];
#pragma unroll
      for (int mi = 0; mi < 2; ++mi) {
        v8h lo  = *(const v8h*)(aPtr[mi] + k0);
        v8h hiv = *(const v8h*)(aPtr[mi] + k0 + 8);
        a[mi] = __builtin_shufflevector(lo, hiv, 0, 1, 2, 3, 4, 5, 6, 7, 8, 9,
                                        10, 11, 12, 13, 14, 15);
      }
      v16h b[4];
#pragma unroll
      for (int j = 0; j < 4; ++j) b[j] = *(const v16h*)(bPtr[j] + k0);
#pragma unroll
      for (int j = 0; j < 4; ++j)
#pragma unroll
        for (int mi = 0; mi < 2; ++mi)
          acc[mi][j] = __builtin_amdgcn_wmma_f32_16x16x32_f16(
              false, a[mi], false, b[j], (short)0, acc[mi][j], false, false);
    }
  } else {
    // -------- edge path: guard each B column --------
    bool bValid[4];
#pragma unroll
    for (int j = 0; j < 4; ++j) bValid[j] = (n0 + j * 16 + r16 < N);
    for (int k0 = 0; k0 < K; k0 += 32) {
      v16h a[2];
#pragma unroll
      for (int mi = 0; mi < 2; ++mi) {
        v8h lo  = *(const v8h*)(aPtr[mi] + k0);
        v8h hiv = *(const v8h*)(aPtr[mi] + k0 + 8);
        a[mi] = __builtin_shufflevector(lo, hiv, 0, 1, 2, 3, 4, 5, 6, 7, 8, 9,
                                        10, 11, 12, 13, 14, 15);
      }
#pragma unroll
      for (int j = 0; j < 4; ++j) {
        v16h b = {};
        if (bValid[j]) b = *(const v16h*)(bPtr[j] + k0);
#pragma unroll
        for (int mi = 0; mi < 2; ++mi)
          acc[mi][j] = __builtin_amdgcn_wmma_f32_16x16x32_f16(
              false, a[mi], false, b, (short)0, acc[mi][j], false, false);
      }
    }
  }

  // Epilogue. C/D layout: element (vgpr r, lane l) = (M = r + 8*(l>>4), N = l&15)
#pragma unroll
  for (int mi = 0; mi < 2; ++mi) {
    const int rowBase = m0 + mi * 16 + hi * 8;
#pragma unroll
    for (int j = 0; j < 4; ++j) {
      const int col = n0 + j * 16 + r16;
      if (fullTile || col < N) {
        const float bval = HAS_BIAS ? bias[col] : 0.0f;
#pragma unroll
        for (int r = 0; r < 8; ++r) {
          float v = acc[mi][j][r] * scale + bval;
          if (DO_GELU) v = gelu_exact(v);
          if (STORE32) C32[(size_t)(rowBase + r) * ldc32 + col] = v;
          if (STORE16) {
            if (TRANS16)
              C16[(size_t)col * ldc16 + (rowBase + r)] = (_Float16)v;
            else
              C16[(size_t)(rowBase + r) * ldc16 + col] = (_Float16)v;
          }
        }
      }
    }
  }
}

// ---------------------------------------------------------------- softmax
// One block per query row; causal: only cols 0..s are live; writes f16 probs.
__global__ void __launch_bounds__(256)
softmax_causal_kernel(const float* __restrict__ scores,
                      _Float16* __restrict__ probs) {
  __shared__ float red[256];
  const int s = blockIdx.x;
  const int t = threadIdx.x;
  const float* rowp = scores + (size_t)s * SEQ;
  _Float16* outp = probs + (size_t)s * SEQ;
  const int n = s + 1;

  float mx = -3.0e38f;
  for (int c = t; c < n; c += 256) mx = fmaxf(mx, rowp[c]);
  red[t] = mx;
  __syncthreads();
  for (int o = 128; o > 0; o >>= 1) {
    if (t < o) red[t] = fmaxf(red[t], red[t + o]);
    __syncthreads();
  }
  mx = red[0];
  __syncthreads();

  float sum = 0.0f;
  for (int c = t; c < n; c += 256) sum += expf(rowp[c] - mx);
  red[t] = sum;
  __syncthreads();
  for (int o = 128; o > 0; o >>= 1) {
    if (t < o) red[t] += red[t + o];
    __syncthreads();
  }
  const float inv = 1.0f / red[0];

  for (int c = t; c < SEQ; c += 256)
    outp[c] = (c < n) ? (_Float16)(expf(rowp[c] - mx) * inv) : (_Float16)0.0f;
}

// ---------------------------------------------------------------- add + LN
// y = LN(x + attn) * g + b, written as f16 (feeds MLP1 GEMM).
__global__ void __launch_bounds__(256)
add_layernorm_kernel(const float* __restrict__ x, const float* __restrict__ at,
                     const float* __restrict__ g, const float* __restrict__ b,
                     _Float16* __restrict__ out) {
  __shared__ float red[256];
  const int s = blockIdx.x;
  const int t = threadIdx.x;
  const float* xr = x + (size_t)s * DMODEL;
  const float* ar = at + (size_t)s * DMODEL;

  float v[3];
  float sum = 0.0f;
#pragma unroll
  for (int i = 0; i < 3; ++i) {
    int d = t + i * 256;
    v[i] = xr[d] + ar[d];
    sum += v[i];
  }
  red[t] = sum;
  __syncthreads();
  for (int o = 128; o > 0; o >>= 1) {
    if (t < o) red[t] += red[t + o];
    __syncthreads();
  }
  const float mu = red[0] * (1.0f / DMODEL);
  __syncthreads();

  float var = 0.0f;
#pragma unroll
  for (int i = 0; i < 3; ++i) {
    float d0 = v[i] - mu;
    var += d0 * d0;
  }
  red[t] = var;
  __syncthreads();
  for (int o = 128; o > 0; o >>= 1) {
    if (t < o) red[t] += red[t + o];
    __syncthreads();
  }
  const float rstd = rsqrtf(red[0] * (1.0f / DMODEL) + 1e-5f);

#pragma unroll
  for (int i = 0; i < 3; ++i) {
    int d = t + i * 256;
    out[(size_t)s * DMODEL + d] = (_Float16)((v[i] - mu) * rstd * g[d] + b[d]);
  }
}

// ---------------------------------------------------------------- launcher
extern "C" void kernel_launch(void* const* d_in, const int* in_sizes, int n_in,
                              void* d_out, int out_size, void* d_ws,
                              size_t ws_size, hipStream_t stream) {
  (void)in_sizes; (void)n_in; (void)out_size; (void)ws_size;
  const int*   tok    = (const int*)d_in[0];
  const float* emb    = (const float*)d_in[1];
  const float* pos    = (const float*)d_in[2];
  const float* wq_w   = (const float*)d_in[3];
  const float* wq_b   = (const float*)d_in[4];
  const float* wk_w   = (const float*)d_in[5];
  const float* wk_b   = (const float*)d_in[6];
  const float* wv_w   = (const float*)d_in[7];
  const float* wv_b   = (const float*)d_in[8];
  const float* ln_g   = (const float*)d_in[9];
  const float* ln_b   = (const float*)d_in[10];
  const float* mlp1_w = (const float*)d_in[11];
  const float* mlp1_b = (const float*)d_in[12];
  const float* mlp2_w = (const float*)d_in[13];
  const float* mlp2_b = (const float*)d_in[14];
  float* out = (float*)d_out;

  // ---- workspace carve-out (256B aligned) ----
  char* base = (char*)d_ws;
  size_t off = 0;
  auto alloc = [&](size_t bytes) -> void* {
    void* p = base + off;
    off += (bytes + 255) & ~(size_t)255;
    return p;
  };
  float*    x32    = (float*)   alloc((size_t)SEQ * DMODEL * 4);
  _Float16* x16    = (_Float16*)alloc((size_t)SEQ * DMODEL * 2);
  _Float16* wq16   = (_Float16*)alloc((size_t)DMODEL * DMODEL * 2);
  _Float16* wk16   = (_Float16*)alloc((size_t)DMODEL * DMODEL * 2);
  _Float16* wv16   = (_Float16*)alloc((size_t)DMODEL * DMODEL * 2);
  _Float16* q16    = (_Float16*)alloc((size_t)SEQ * DMODEL * 2);
  _Float16* k16    = (_Float16*)alloc((size_t)SEQ * DMODEL * 2);
  _Float16* vt16   = (_Float16*)alloc((size_t)DMODEL * SEQ * 2);  // [D, S]
  float*    scores = (float*)   alloc((size_t)SEQ * SEQ * 4);     // one head
  _Float16* probs  = (_Float16*)alloc((size_t)SEQ * SEQ * 2);     // one head
  float*    attn32 = (float*)   alloc((size_t)SEQ * DMODEL * 4);
  _Float16* lnout  = (_Float16*)alloc((size_t)SEQ * DMODEL * 2);
  _Float16* w1_16  = (_Float16*)alloc((size_t)DFF * DMODEL * 2);
  _Float16* h16    = (_Float16*)alloc((size_t)SEQ * DFF * 2);
  _Float16* w2_16  = (_Float16*)alloc((size_t)DMODEL * DFF * 2);
  _Float16* x2_16  = (_Float16*)alloc((size_t)SEQ * DMODEL * 2);
  _Float16* emb16  = (_Float16*)alloc((size_t)NVOCAB * DMODEL * 2);

  auto cvt = [&](const float* src, _Float16* dst, size_t n) {
    f32_to_f16_kernel<<<dim3((unsigned)((n + 255) / 256)), 256, 0, stream>>>(
        src, dst, (int)n);
  };

  // ---- embedding + weight conversion ----
  embed_kernel<<<dim3(SEQ * DMODEL / 256), 256, 0, stream>>>(tok, emb, pos,
                                                             x32, x16);
  cvt(wq_w, wq16, (size_t)DMODEL * DMODEL);
  cvt(wk_w, wk16, (size_t)DMODEL * DMODEL);
  cvt(wv_w, wv16, (size_t)DMODEL * DMODEL);
  cvt(mlp1_w, w1_16, (size_t)DFF * DMODEL);
  cvt(mlp2_w, w2_16, (size_t)DMODEL * DFF);
  cvt(emb, emb16, (size_t)NVOCAB * DMODEL);

  const dim3 blk(32);

  // ---- QKV projections: [S,768] = x @ w^T + b ----
  // q, k: plain f16 store. v: store transposed so each head's block
  // vt16[h*HD ..] is [HD, S] row-major => Bt form for the attn GEMM.
  wmma_gemm<true, false, false, true, false, false>
      <<<dim3(DMODEL / 64, SEQ / 32), blk, 0, stream>>>(
          x16, DMODEL, wq16, DMODEL, wq_b, nullptr, 0, q16, DMODEL, DMODEL,
          DMODEL, 1.0f);
  wmma_gemm<true, false, false, true, false, false>
      <<<dim3(DMODEL / 64, SEQ / 32), blk, 0, stream>>>(
          x16, DMODEL, wk16, DMODEL, wk_b, nullptr, 0, k16, DMODEL, DMODEL,
          DMODEL, 1.0f);
  wmma_gemm<true, false, false, true, true, false>
      <<<dim3(DMODEL / 64, SEQ / 32), blk, 0, stream>>>(
          x16, DMODEL, wv16, DMODEL, wv_b, nullptr, 0, vt16, SEQ, DMODEL,
          DMODEL, 1.0f);

  // ---- attention, one head at a time (stream-ordered) ----
  const float scl = 0.10206207261596576f;  // 1/sqrt(96)
  for (int h = 0; h < NHEADS; ++h) {
    // scores = (q_h @ k_h^T) / sqrt(hd), causal tiles skipped
    wmma_gemm<false, false, true, false, false, true>
        <<<dim3(SEQ / 64, SEQ / 32), blk, 0, stream>>>(
            q16 + h * HDIM, DMODEL, k16 + h * HDIM, DMODEL, nullptr, scores,
            SEQ, nullptr, 0, SEQ, HDIM, scl);
    softmax_causal_kernel<<<dim3(SEQ), 256, 0, stream>>>(scores, probs);
    // attn_h = probs @ v_h  (v stored transposed: Bt = vt16[h*HD ..][S])
    wmma_gemm<false, false, true, false, false, false>
        <<<dim3((HDIM + 63) / 64, SEQ / 32), blk, 0, stream>>>(
            probs, SEQ, vt16 + (size_t)h * HDIM * SEQ, SEQ, nullptr,
            attn32 + h * HDIM, DMODEL, nullptr, 0, HDIM, SEQ, 1.0f);
  }

  // ---- residual + LayerNorm ----
  add_layernorm_kernel<<<dim3(SEQ), 256, 0, stream>>>(x32, attn32, ln_g, ln_b,
                                                      lnout);

  // ---- MLP ----
  wmma_gemm<true, true, false, true, false, false>
      <<<dim3(DFF / 64, SEQ / 32), blk, 0, stream>>>(
          lnout, DMODEL, w1_16, DMODEL, mlp1_b, nullptr, 0, h16, DFF, DFF,
          DMODEL, 1.0f);
  wmma_gemm<true, false, false, true, false, false>
      <<<dim3(DMODEL / 64, SEQ / 32), blk, 0, stream>>>(
          h16, DFF, w2_16, DFF, mlp2_b, nullptr, 0, x2_16, DMODEL, DMODEL, DFF,
          1.0f);

  // ---- LM head (weight-tied): out[S,V] = x2 @ in_embed^T ----
  wmma_gemm<false, false, true, false, false, false>
      <<<dim3((NVOCAB + 63) / 64, SEQ / 32), blk, 0, stream>>>(
          x2_16, DMODEL, emb16, DMODEL, nullptr, out, NVOCAB, nullptr, 0,
          NVOCAB, DMODEL, 1.0f);
}
